// ContextLearner_29283087024281
// MI455X (gfx1250) — matrix operand order repaired and checked by reading, not compiled, not run on hardware
//
#include <hip/hip_runtime.h>
#include <hip/hip_bf16.h>

typedef __attribute__((ext_vector_type(16))) _Float16 v16h;
typedef __attribute__((ext_vector_type(8)))  _Float16 v8h;
typedef __attribute__((ext_vector_type(8)))  float    v8f;

namespace {
constexpr int BN_TOT = 1024;   // 512 * 2
constexpr int NN     = 100;    // neighbors
constexpr int DIM    = 512;
constexpr int HEADS  = 4;
constexpr int HD     = 128;
constexpr int MP     = 112;    // 7 tiles of 16 (row padding)
constexpr int KP     = 128;    // padded key/col dim for attention
constexpr int QKVC   = 1536;

// ---- LDS layout (bytes) ----
constexpr int OFF_XNH = 0;                    // f16 [100][512]
constexpr int SZ_XNH  = NN * DIM * 2;
constexpr int OFF_QH  = OFF_XNH + SZ_XNH;     // f16 [112][128]
constexpr int SZ_QH   = MP * HD * 2;
constexpr int OFF_KHT = OFF_QH + SZ_QH;       // f16 [128][112]  (k transposed)
constexpr int SZ_KHT  = HD * MP * 2;
constexpr int OFF_VH  = OFF_KHT + SZ_KHT;     // f16 [112][128]
constexpr int SZ_VH   = MP * HD * 2;
constexpr int OFF_ATT = OFF_VH + SZ_VH;       // f16 [112][128]
constexpr int SZ_ATT  = MP * KP * 2;
constexpr int OFF_Y   = OFF_ATT + SZ_ATT;     // f16 [100][512]
constexpr int SZ_Y    = NN * DIM * 2;
constexpr int OFF_AM  = OFF_Y + SZ_Y;         // f32 [4][128]   attn col means
constexpr int SZ_AM   = HEADS * KP * 4;
constexpr int OFF_AMP = OFF_AM + SZ_AM;       // f32 [8][128]   per-wave partials
constexpr int SZ_AMP  = 8 * KP * 4;
constexpr int OFF_Z   = OFF_AMP + SZ_AMP;     // 64B zero scratch for OOB tiles
constexpr int SZ_Z    = 64;
constexpr int SMEM_BYTES = OFF_Z + SZ_Z;      // 325,696 <= 327,680 (320KB WGP)
constexpr int OFF_WC  = OFF_QH;               // f32 [4][512] overlays freed qh

// element (half) offsets into smem
constexpr int XNH_E = OFF_XNH / 2;
constexpr int QH_E  = OFF_QH  / 2;
constexpr int KHT_E = OFF_KHT / 2;
constexpr int VH_E  = OFF_VH  / 2;
constexpr int ATT_E = OFF_ATT / 2;
constexpr int Y_E   = OFF_Y   / 2;
constexpr int Z_E   = OFF_Z   / 2;
}

#define WMMA_F16(a, b, c) \
  __builtin_amdgcn_wmma_f32_16x16x32_f16(false, (a), false, (b), (short)0, (c), false, false)

__device__ __forceinline__ v16h cat8(v8h lo, v8h hv) {
  return __builtin_shufflevector(lo, hv, 0, 1, 2, 3, 4, 5, 6, 7,
                                         8, 9, 10, 11, 12, 13, 14, 15);
}

// ---- A tile (16x32 f16, M x K) from LDS, row-major, stride in halfs.
// Lanes0-15: M=0..15, elems {K0..7, K16..23}; lanes16-31 same M, {K8..15, K24..31}.
// Branch-free OOB: rows >= maxRow redirect to a zeroed LDS scratch (v_cndmask).
__device__ __forceinline__ v16h ldA(const _Float16* sm, int matOff, int stride,
                                    int mB, int kB, int maxRow, int l15, int hi) {
  int row = mB + l15;
  int off = (row < maxRow) ? (matOff + row * stride + kB + hi * 8) : Z_E;
  const _Float16* p = sm + off;
  v8h lo = *(const v8h*)p;        // K g=0 chunk (16B aligned)
  v8h hv = *(const v8h*)(p + 16); // K g=1 chunk
  return cat8(lo, hv);
}

// ---- B tile (32x16 f16, K x N) from LDS K-major matrix: lane=N col, elems {K0..7,K16..23} / {K8..15,K24..31}.
__device__ __forceinline__ v16h ldBs(const _Float16* sm, int matOff, int stride,
                                     int kB, int nB, int lane) {
  v16h r;
#pragma unroll
  for (int s = 0; s < 16; ++s) {
    int k = kB + (lane >> 4) * 8 + (s < 8 ? s : s + 8);
    r[s] = sm[matOff + k * stride + nB + (lane & 15)];
  }
  return r;
}
// Guarded variant: K rows >= maxK read zero.
__device__ __forceinline__ v16h ldBsg(const _Float16* sm, int matOff, int stride,
                                      int kB, int nB, int maxK, int lane) {
  v16h r;
#pragma unroll
  for (int s = 0; s < 16; ++s) {
    int k = kB + (lane >> 4) * 8 + (s < 8 ? s : s + 8);
    int off = (k < maxK) ? (matOff + k * stride + nB + (lane & 15)) : Z_E;
    r[s] = sm[off];
  }
  return r;
}
// ---- B tile from global K-major f16 matrix (lane=N col, strided K gather).
__device__ __forceinline__ v16h ldBgm(const _Float16* base, int stride, int kB,
                                      int nB, int lane) {
  v16h r;
#pragma unroll
  for (int s = 0; s < 16; ++s) {
    int k = kB + (lane >> 4) * 8 + (s < 8 ? s : s + 8);
    r[s] = base[k * stride + nB + (lane & 15)];
  }
  return r;
}

__device__ __forceinline__ float wredSum(float v) {
#pragma unroll
  for (int m = 16; m > 0; m >>= 1) v += __shfl_xor(v, m, 32);
  return v;
}
__device__ __forceinline__ float wredMax(float v) {
#pragma unroll
  for (int m = 16; m > 0; m >>= 1) v = fmaxf(v, __shfl_xor(v, m, 32));
  return v;
}

// ---- weight prep: f32 row-major [N][K] -> f16 K-major [K][N] in workspace ----
__global__ void prep_weights_kernel(const float* __restrict__ qkv_w,
                                    const float* __restrict__ proj_w,
                                    _Float16* __restrict__ qkv_wt,
                                    _Float16* __restrict__ proj_wt) {
  int i = blockIdx.x * 256 + threadIdx.x;   // grid covers 786432
  if (i < 512 * QKVC) {                     // qkv_wt[d][j] = qkv_w[j][d]
    int d = i / QKVC, j = i - d * QKVC;
    qkv_wt[i] = (_Float16)qkv_w[j * 512 + d];
  }
  if (i < 512 * 512) {                      // proj_wt[d][j] = proj_w[j][d]
    int d = i >> 9, j = i & 511;
    proj_wt[i] = (_Float16)proj_w[j * 512 + d];
  }
}

__global__ void __launch_bounds__(256)
context_learner_kernel(const int* __restrict__ conn, const int* __restrict__ maskp,
                       const float* __restrict__ emb,
                       const float* __restrict__ qkv_b, const float* __restrict__ proj_b,
                       const float* __restrict__ ln1_g, const float* __restrict__ ln1_b,
                       const float* __restrict__ ln2_g, const float* __restrict__ ln2_b,
                       const float* __restrict__ fc_w, const float* __restrict__ fc_b,
                       const _Float16* __restrict__ qkv_wt,
                       const _Float16* __restrict__ proj_wt,
                       float* __restrict__ out) {
  extern __shared__ char smem[];
  _Float16* sm  = (_Float16*)smem;          // half-typed base for tile loaders
  _Float16* xnh = (_Float16*)(smem + OFF_XNH);
  _Float16* qh  = (_Float16*)(smem + OFF_QH);
  _Float16* khT = (_Float16*)(smem + OFF_KHT);
  _Float16* vh  = (_Float16*)(smem + OFF_VH);
  _Float16* att = (_Float16*)(smem + OFF_ATT);
  _Float16* yld = (_Float16*)(smem + OFF_Y);
  float*    am  = (float*)(smem + OFF_AM);
  float*    amp = (float*)(smem + OFF_AMP);
  float*    wc  = (float*)(smem + OFF_WC);

  const int bn   = blockIdx.x;
  const int tid  = threadIdx.x;
  const int wave = tid >> 5;
  const int lane = tid & 31;
  const int l15  = lane & 15;
  const int hi   = lane >> 4;

  // zero scratch for branch-free OOB tile reads
  if (tid < 32) ((_Float16*)(smem + OFF_Z))[tid] = (_Float16)0.f;

  // ---------- Phase 0: gather two embeddings per row, LayerNorm1, f16 to LDS ----------
  {
    float gv[16], bv[16];
    const int cbase = lane * 16;
#pragma unroll
    for (int i = 0; i < 16; ++i) { gv[i] = ln1_g[cbase + i]; bv[i] = ln1_b[cbase + i]; }
    for (int row = wave; row < NN; row += 8) {
      const int* cp = conn + (bn * NN + row) * 2;
      int sym = hi ? cp[1] : cp[0];
      const float* src = emb + (long)sym * 256 + l15 * 16;
      float v[16];
#pragma unroll
      for (int i = 0; i < 16; i += 4) {
        float4 f = *(const float4*)(src + i);
        v[i] = f.x; v[i + 1] = f.y; v[i + 2] = f.z; v[i + 3] = f.w;
      }
      float s = 0.f, sq = 0.f;
#pragma unroll
      for (int i = 0; i < 16; ++i) { s += v[i]; sq += v[i] * v[i]; }
      s = wredSum(s); sq = wredSum(sq);
      float mean = s * (1.f / 512.f);
      float var  = sq * (1.f / 512.f) - mean * mean;
      float rs   = rsqrtf(var + 1e-5f);
      _Float16* dst = xnh + row * DIM + cbase;
#pragma unroll
      for (int i = 0; i < 16; ++i)
        dst[i] = (_Float16)((v[i] - mean) * rs * gv[i] + bv[i]);
    }
  }
  __syncthreads();

  const float scale = 0.0883883476483184f;  // 1/sqrt(128)

  // ---------- Per-head attention ----------
  for (int h = 0; h < HEADS; ++h) {
    // ---- QKV GEMM: 3 mats x 7 Mtiles x 8 Ntiles, K=512 (16 steps, dbl-buffered) ----
    for (int t = wave; t < 168; t += 8) {
      int mat = t / 56, rem = t - mat * 56, mt = rem >> 3, nt = rem & 7;
      int colBase = mat * DIM + h * HD + nt * 16;
      v8f acc = {};
      v16h a0 = ldA(sm, XNH_E, DIM, mt * 16, 0, NN, l15, hi);
      v16h b0 = ldBgm(qkv_wt, QKVC, 0, colBase, lane);
      for (int kt = 0; kt < 15; ++kt) {
        v16h a1 = ldA(sm, XNH_E, DIM, mt * 16, (kt + 1) * 32, NN, l15, hi);
        v16h b1 = ldBgm(qkv_wt, QKVC, (kt + 1) * 32, colBase, lane);
        acc = WMMA_F16(a0, b0, acc);
        a0 = a1; b0 = b1;
      }
      acc = WMMA_F16(a0, b0, acc);
      float bias = qkv_b[colBase + l15];
      int row0 = mt * 16 + hi * 8;
      int d = nt * 16 + l15;
#pragma unroll
      for (int r = 0; r < 8; ++r) {
        float vv = acc[r] + bias;
        int row = row0 + r;
        if (mat == 0)      qh[row * HD + d]  = (_Float16)vv;
        else if (mat == 1) khT[d * MP + row] = (_Float16)vv;   // transposed store
        else               vh[row * HD + d]  = (_Float16)vv;
      }
    }
    __syncthreads();

    // ---- pad attn cols 112..127 with large-negative ----
    for (int i = tid; i < MP * 16; i += 256) {
      int r = i >> 4, c = 112 + (i & 15);
      att[r * KP + c] = (_Float16)(-60000.f);
    }
    // ---- scores S = q k^T * scale, mask, write f16 logits ----
    for (int t = wave; t < 49; t += 8) {
      int mt = t / 7, nt = t - mt * 7;
      v8f acc = {};
      v16h a0 = ldA(sm, QH_E, HD, mt * 16, 0, MP, l15, hi);
      v16h b0 = ldBs(sm, KHT_E, MP, 0, nt * 16, lane);
      for (int kt = 0; kt < 3; ++kt) {
        v16h a1 = ldA(sm, QH_E, HD, mt * 16, (kt + 1) * 32, MP, l15, hi);
        v16h b1 = ldBs(sm, KHT_E, MP, (kt + 1) * 32, nt * 16, lane);
        acc = WMMA_F16(a0, b0, acc);
        a0 = a1; b0 = b1;
      }
      acc = WMMA_F16(a0, b0, acc);
      int q0 = mt * 16 + hi * 8, key = nt * 16 + l15;
#pragma unroll
      for (int r = 0; r < 8; ++r) {
        int qrow = q0 + r;
        float logit;
        if (qrow < NN) {
          if (key < NN) {
            int m = maskp[bn * (NN * NN) + qrow * NN + key];
            logit = m ? acc[r] * scale : -30000.f;  // finite, behaves like -1e9
          } else {
            logit = -60000.f;                       // padded keys -> prob exactly 0
          }
        } else {
          logit = 0.f;                              // padded query rows: benign
        }
        att[qrow * KP + key] = (_Float16)logit;
      }
    }
    __syncthreads();

    // ---- softmax per row + per-wave attn-mean partials ----
    {
      float amr[4] = {0.f, 0.f, 0.f, 0.f};
      for (int row = wave; row < MP; row += 8) {
        _Float16* ar = att + row * KP + lane * 4;
        float v[4];
#pragma unroll
        for (int i = 0; i < 4; ++i) v[i] = (float)ar[i];
        float mx = fmaxf(fmaxf(v[0], v[1]), fmaxf(v[2], v[3]));
        mx = wredMax(mx);
        float e[4], s = 0.f;
#pragma unroll
        for (int i = 0; i < 4; ++i) { e[i] = __expf(v[i] - mx); s += e[i]; }
        s = wredSum(s);
        float inv = 1.f / s;
#pragma unroll
        for (int i = 0; i < 4; ++i) {
          float p = e[i] * inv;
          ar[i] = (_Float16)p;
          if (row < NN) amr[i] += p;
        }
      }
#pragma unroll
      for (int i = 0; i < 4; ++i) amp[wave * KP + lane * 4 + i] = amr[i];
    }
    __syncthreads();
    if (tid < KP) {  // reduce 8 wave partials -> mean over the 100 query rows
      float s = 0.f;
#pragma unroll
      for (int w = 0; w < 8; ++w) s += amp[w * KP + tid];
      am[h * KP + tid] = s * 0.01f;
    }

    // ---- y_head = attn @ v  (K = 128 padded keys; v rows >=112 read zeros) ----
    for (int t = wave; t < 56; t += 8) {
      int mt = t >> 3, nt = t & 7;
      v8f acc = {};
      v16h a0 = ldA(sm, ATT_E, KP, mt * 16, 0, MP, l15, hi);
      v16h b0 = ldBsg(sm, VH_E, HD, 0, nt * 16, MP, lane);
      for (int kt = 0; kt < 3; ++kt) {
        v16h a1 = ldA(sm, ATT_E, KP, mt * 16, (kt + 1) * 32, MP, l15, hi);
        v16h b1 = ldBsg(sm, VH_E, HD, (kt + 1) * 32, nt * 16, MP, lane);
        acc = WMMA_F16(a0, b0, acc);
        a0 = a1; b0 = b1;
      }
      acc = WMMA_F16(a0, b0, acc);
      int row0 = mt * 16 + hi * 8, d = nt * 16 + l15;
#pragma unroll
      for (int r = 0; r < 8; ++r) {
        int row = row0 + r;
        if (row < NN) yld[row * DIM + h * HD + d] = (_Float16)acc[r];
      }
    }
    __syncthreads();
  }

  // ---------- proj + wc: wc[h,c] = sum_n am[h,n] * (y @ proj_w^T + b)[n,c] ----------
  for (int ct = wave; ct < 32; ct += 8) {       // wave owns column tiles ct,ct+8,...
    float wcp[4] = {0.f, 0.f, 0.f, 0.f};
    for (int nt = 0; nt < 7; ++nt) {
      v8f acc = {};
      v16h a0 = ldA(sm, Y_E, DIM, nt * 16, 0, NN, l15, hi);
      v16h b0 = ldBgm(proj_wt, DIM, 0, ct * 16, lane);
      for (int kt = 0; kt < 15; ++kt) {
        v16h a1 = ldA(sm, Y_E, DIM, nt * 16, (kt + 1) * 32, NN, l15, hi);
        v16h b1 = ldBgm(proj_wt, DIM, (kt + 1) * 32, ct * 16, lane);
        acc = WMMA_F16(a0, b0, acc);
        a0 = a1; b0 = b1;
      }
      acc = WMMA_F16(a0, b0, acc);
      int c = ct * 16 + l15;
      float pb = proj_b[c];
      int row0 = nt * 16 + hi * 8;
#pragma unroll
      for (int r = 0; r < 8; ++r) {
        int row = row0 + r;
        if (row < NN) {
          float val = acc[r] + pb;
#pragma unroll
          for (int hh = 0; hh < 4; ++hh) wcp[hh] += am[hh * KP + row] * val;
        }
      }
    }
#pragma unroll
    for (int hh = 0; hh < 4; ++hh) {
      wcp[hh] += __shfl_xor(wcp[hh], 16, 32);   // merge hi/lo half-lane rows
      if (hi == 0) wc[hh * DIM + ct * 16 + l15] = wcp[hh];
    }
  }
  __syncthreads();

  // ---------- LayerNorm2 over wc rows (one wave per head) ----------
  if (wave < HEADS) {
    float* wr = wc + wave * DIM + lane * 16;
    float v[16];
#pragma unroll
    for (int i = 0; i < 16; ++i) v[i] = wr[i];
    float s = 0.f, sq = 0.f;
#pragma unroll
    for (int i = 0; i < 16; ++i) { s += v[i]; sq += v[i] * v[i]; }
    s = wredSum(s); sq = wredSum(sq);
    float mean = s * (1.f / 512.f);
    float var  = sq * (1.f / 512.f) - mean * mean;
    float rs   = rsqrtf(var + 1e-5f);
    int cbase = lane * 16;
#pragma unroll
    for (int i = 0; i < 16; ++i)
      wr[i] = (v[i] - mean) * rs * ln2_g[cbase + i] + ln2_b[cbase + i];
  }
  __syncthreads();

  // ---------- fc: out[h, j] = ln(wc)[h] . fc_w[j] + fc_b[j] ----------
  {
    int j = tid;                                 // 0..255
    const float* wrow = fc_w + j * DIM;
    float acc[4] = {0.f, 0.f, 0.f, 0.f};
    for (int d0 = 0; d0 < DIM; d0 += 4) {
      float4 w4 = *(const float4*)(wrow + d0);
#pragma unroll
      for (int hh = 0; hh < 4; ++hh) {
        const float* wch = wc + hh * DIM + d0;
        acc[hh] += wch[0] * w4.x + wch[1] * w4.y + wch[2] * w4.z + wch[3] * w4.w;
      }
    }
    float fb = fc_b[j];
#pragma unroll
    for (int hh = 0; hh < 4; ++hh)
      out[(long)bn * (HEADS * 256) + hh * 256 + j] = acc[hh] + fb;
  }
}

extern "C" void kernel_launch(void* const* d_in, const int* in_sizes, int n_in,
                              void* d_out, int out_size, void* d_ws, size_t ws_size,
                              hipStream_t stream) {
  (void)in_sizes; (void)n_in; (void)out_size; (void)ws_size;
  const int*   conn   = (const int*)  d_in[0];
  const int*   maskp  = (const int*)  d_in[1];
  const float* emb    = (const float*)d_in[2];
  const float* qkv_w  = (const float*)d_in[3];
  const float* qkv_b  = (const float*)d_in[4];
  const float* proj_w = (const float*)d_in[5];
  const float* proj_b = (const float*)d_in[6];
  const float* ln1_g  = (const float*)d_in[7];
  const float* ln1_b  = (const float*)d_in[8];
  const float* ln2_g  = (const float*)d_in[9];
  const float* ln2_b  = (const float*)d_in[10];
  const float* fc_w   = (const float*)d_in[11];
  const float* fc_b   = (const float*)d_in[12];
  float* out = (float*)d_out;

  _Float16* qkv_wt  = (_Float16*)d_ws;                             // 512x1536 f16
  _Float16* proj_wt = (_Float16*)((char*)d_ws + 512 * 1536 * 2);   // 512x512  f16

  (void)hipFuncSetAttribute((const void*)context_learner_kernel,
                            hipFuncAttributeMaxDynamicSharedMemorySize, SMEM_BYTES);

  prep_weights_kernel<<<3072, 256, 0, stream>>>(qkv_w, proj_w, qkv_wt, proj_wt);
  context_learner_kernel<<<BN_TOT, 256, SMEM_BYTES, stream>>>(
      conn, maskp, emb, qkv_b, proj_b, ln1_g, ln1_b, ln2_g, ln2_b,
      fc_w, fc_b, qkv_wt, proj_wt, out);
}
